// BiLstmCellEncoder_40209483825496
// MI455X (gfx1250) — compile-verified, matrix operand order checked by reading, not run on hardware
//
#include <hip/hip_runtime.h>
#include <hip/hip_bf16.h>
#include <stdint.h>

// ---------------------------------------------------------------------------
// BiLSTM (fw+bw) + self-attention for MI455X (gfx1250), wave32 + WMMA bf16.
// B=32, S=1024, I=H=128, D=256.
// ---------------------------------------------------------------------------

typedef __attribute__((ext_vector_type(16))) __bf16 v16bf;
typedef __attribute__((ext_vector_type(8)))  float  v8f;

#define NB 32        // batch
#define NS 1024      // seq len
#define NI 128       // input dim
#define NH 128       // hidden
#define ND 256       // 2*H
#define NG 512       // 4*H

// ----- workspace layout (bytes) -----
#define OFF_XG_FW  ((size_t)0x0000000)   // [B*S][512] f32  (dead after LSTM)
#define OFF_XG_BW  ((size_t)0x4000000)   // [B*S][512] f32  (dead after LSTM)
#define OFF_H      ((size_t)0x8000000)   // [B*S][256] bf16
#define OFF_XBF    ((size_t)0x9000000)   // [B*S][128] bf16
#define OFF_XREV   ((size_t)0x9800000)   // [B*S][128] bf16 (seq reversed)
#define OFF_WTS    ((size_t)0xA000000)   // packed bf16 weights + f32 biases
#define OFF_Q      ((size_t)0x0000000)   // overlay dead xg_fw
#define OFF_K      ((size_t)0x1000000)
#define OFF_VT     ((size_t)0x2000000)   // V stored transposed per batch [b][n][s]

// ---------------------------------------------------------------------------
// WMMA helpers (ISA 7.12.2 layouts).
// 16-bit A 16x32: lane = {half=lane>>4, row=lane&15}; VGPR j holds K pair at
// kbase = half*8 + 2*(j&3) + ((j>>2)<<4).
// ---------------------------------------------------------------------------
__device__ inline v8f wmma_bf16(v16bf a, v16bf b, v8f c) {
  return __builtin_amdgcn_wmma_f32_16x16x32_bf16(
      /*neg_a=*/false, a, /*neg_b=*/false, b,
      /*c_mod=*/(short)0, c, /*reuse_a=*/false, /*reuse_b=*/false);
}

// A-tile (16x32) from row-major [16 x ld] bf16; also B-tile when the weight
// matrix is stored as [N x K] row-major (i.e. B = W^T). Consecutive K pairs
// let the compiler fuse lanes' gathers into global_load_b128.
__device__ inline v16bf ld_tile_rowmaj(const __bf16* base, int ld, int lane) {
  const int hf = lane >> 4, row = lane & 15;
  const __bf16* p = base + row * ld;
  v16bf r;
#pragma unroll
  for (int j = 0; j < 8; ++j) {
    const int kb = hf * 8 + 2 * (j & 3) + ((j >> 2) << 4);
    r[2 * j]     = p[kb];
    r[2 * j + 1] = p[kb + 1];
  }
  return r;
}

__device__ inline float sigf(float x)   { return 1.0f / (1.0f + __expf(-x)); }
__device__ inline float tanhf2(float x) { return 2.0f * sigf(2.0f * x) - 1.0f; }

// ---------------------------------------------------------------------------
// Packing kernels
// ---------------------------------------------------------------------------
__global__ void k_cvt_bf16(const float* __restrict__ s, __bf16* __restrict__ d, int n) {
  int i = blockIdx.x * blockDim.x + threadIdx.x;
  if (i < n) d[i] = (__bf16)s[i];
}

__global__ void k_rev_x(const __bf16* __restrict__ s, __bf16* __restrict__ d) {
  int i = blockIdx.x * blockDim.x + threadIdx.x;   // n = B*S*I = 2^22
  int c = i & 127;
  int t = (i >> 7) & 1023;
  int b = i >> 17;
  d[i] = s[(b << 17) + ((1023 - t) << 7) + c];
}

__global__ void k_bias_add(const float* __restrict__ a, const float* __restrict__ b,
                           float* __restrict__ o, int n) {
  int i = blockIdx.x * blockDim.x + threadIdx.x;
  if (i < n) o[i] = a[i] + b[i];
}

// ---------------------------------------------------------------------------
// xg = X @ Wih^T + bias  : M=B*S, N=512, K=128, fp32 out.
// block = 256 thr (8 waves), tile 64x64, 2 C-tiles/wave, 4 K-steps.
// ---------------------------------------------------------------------------
__global__ void k_gemm_xg(const __bf16* __restrict__ X, const __bf16* __restrict__ W,
                          const float* __restrict__ bias, float* __restrict__ out) {
  const int m0 = blockIdx.x * 64, n0 = blockIdx.y * 64;
  const int wv = threadIdx.x >> 5, lane = threadIdx.x & 31;
  const int t0 = wv * 2;
  const int mt = t0 >> 2, ntA = t0 & 3, ntB = ntA + 1;
  v8f acc0 = {}, acc1 = {};
#pragma unroll
  for (int kc = 0; kc < 4; ++kc) {
    v16bf a  = ld_tile_rowmaj(X + (m0 + mt * 16) * NI + kc * 32, NI, lane);
    v16bf bA = ld_tile_rowmaj(W + (n0 + ntA * 16) * NI + kc * 32, NI, lane);
    v16bf bB = ld_tile_rowmaj(W + (n0 + ntB * 16) * NI + kc * 32, NI, lane);
    acc0 = wmma_bf16(a, bA, acc0);
    acc1 = wmma_bf16(a, bB, acc1);
  }
  const int hf = lane >> 4, cl = lane & 15;
#pragma unroll
  for (int r = 0; r < 8; ++r) {
    const int row = m0 + mt * 16 + r + 8 * hf;
    const int cA = n0 + ntA * 16 + cl;
    const int cB = n0 + ntB * 16 + cl;
    out[row * NG + cA] = acc0[r] + bias[cA];
    out[row * NG + cB] = acc1[r] + bias[cB];
  }
}

// ---------------------------------------------------------------------------
// Persistent LSTM recurrence. grid=2 (dir), block=1024 (32 waves).
// Wave w owns gate columns [16w,16w+16): 2 C-tiles (M=32), Whh slice preloaded
// into VGPRs (4 x v16bf). h (bf16) and c (f32) live in LDS; gates staged in
// LDS for the fused elementwise phase. Dyn LDS = 64K + 16K + 8K = 88K.
// ---------------------------------------------------------------------------
__global__ void k_lstm(const float* __restrict__ xg_fw, const float* __restrict__ xg_bw,
                       const __bf16* __restrict__ whh_fw, const __bf16* __restrict__ whh_bw,
                       __bf16* __restrict__ h) {
  extern __shared__ char smem[];
  float*  gates = (float*)smem;                       // [32][512]
  float*  cbuf  = (float*)(smem + 65536);             // [32][128]
  __bf16* hbuf  = (__bf16*)(smem + 65536 + 16384);    // [32][128]

  const int dir = blockIdx.x;
  const float*  xg = dir ? xg_bw : xg_fw;
  const __bf16* W  = dir ? whh_bw : whh_fw;
  const int tid = threadIdx.x, wv = tid >> 5, lane = tid & 31;

  for (int u = tid; u < NB * NH; u += 1024) { cbuf[u] = 0.0f; hbuf[u] = (__bf16)0.0f; }
  __syncthreads();

  const int n0 = wv * 16;
  v16bf Breg[4];
#pragma unroll
  for (int kc = 0; kc < 4; ++kc)
    Breg[kc] = ld_tile_rowmaj(W + n0 * NH + kc * 32, NH, lane);

  const int hf = lane >> 4, cl = lane & 15;
  const int col = n0 + cl;

  for (int t = 0; t < NS; ++t) {
    v8f acc0 = {}, acc1 = {};
#pragma unroll
    for (int kc = 0; kc < 4; ++kc) {
      v16bf a0 = ld_tile_rowmaj(hbuf + kc * 32, NH, lane);            // rows 0..15
      v16bf a1 = ld_tile_rowmaj(hbuf + 16 * NH + kc * 32, NH, lane);  // rows 16..31
      acc0 = wmma_bf16(a0, Breg[kc], acc0);
      acc1 = wmma_bf16(a1, Breg[kc], acc1);
    }
#pragma unroll
    for (int r = 0; r < 8; ++r) {
      const int m0 = r + 8 * hf, m1 = 16 + r + 8 * hf;
      gates[m0 * NG + col] = acc0[r] + xg[(m0 * NS + t) * NG + col];
      gates[m1 * NG + col] = acc1[r] + xg[(m1 * NS + t) * NG + col];
    }
    __syncthreads();
    for (int u = tid; u < NB * NH; u += 1024) {
      const int m = u >> 7, hx = u & 127;
      const float iv = sigf(gates[m * NG + hx]);
      const float fv = sigf(gates[m * NG + NH + hx]);
      const float gv = tanhf2(gates[m * NG + 2 * NH + hx]);
      const float ov = sigf(gates[m * NG + 3 * NH + hx]);
      const float c  = fv * cbuf[u] + iv * gv;
      const float hv = ov * tanhf2(c);
      cbuf[u] = c;
      const __bf16 hb = (__bf16)hv;
      hbuf[u] = hb;
      const int s = dir ? (NS - 1 - t) : t;
      h[(m * NS + s) * ND + dir * NH + hx] = hb;
    }
    __syncthreads();
  }
}

// ---------------------------------------------------------------------------
// q/k = h @ W^T : M=B*S, N=256, K=256, bf16 out row-major.
// ---------------------------------------------------------------------------
__global__ void k_gemm_qk(const __bf16* __restrict__ H, const __bf16* __restrict__ W,
                          __bf16* __restrict__ out) {
  const int m0 = blockIdx.x * 64, n0 = blockIdx.y * 64;
  const int wv = threadIdx.x >> 5, lane = threadIdx.x & 31;
  const int t0 = wv * 2;
  const int mt = t0 >> 2, ntA = t0 & 3, ntB = ntA + 1;
  v8f acc0 = {}, acc1 = {};
#pragma unroll
  for (int kc = 0; kc < 8; ++kc) {
    v16bf a  = ld_tile_rowmaj(H + (m0 + mt * 16) * ND + kc * 32, ND, lane);
    v16bf bA = ld_tile_rowmaj(W + (n0 + ntA * 16) * ND + kc * 32, ND, lane);
    v16bf bB = ld_tile_rowmaj(W + (n0 + ntB * 16) * ND + kc * 32, ND, lane);
    acc0 = wmma_bf16(a, bA, acc0);
    acc1 = wmma_bf16(a, bB, acc1);
  }
  const int hf = lane >> 4, cl = lane & 15;
#pragma unroll
  for (int r = 0; r < 8; ++r) {
    const int row = m0 + mt * 16 + r + 8 * hf;
    out[row * ND + n0 + ntA * 16 + cl] = (__bf16)acc0[r];
    out[row * ND + n0 + ntB * 16 + cl] = (__bf16)acc1[r];
  }
}

// ---------------------------------------------------------------------------
// v = h @ Wv^T, stored TRANSPOSED per batch: vT[b][n][s]  (n in [0,256)).
// One-time strided stores here turn the flash-attention P@V B-operand into
// coalescable row-major b128 loads.
// ---------------------------------------------------------------------------
__global__ void k_gemm_vT(const __bf16* __restrict__ H, const __bf16* __restrict__ W,
                          __bf16* __restrict__ vT) {
  const int m0 = blockIdx.x * 64, n0 = blockIdx.y * 64;
  const int wv = threadIdx.x >> 5, lane = threadIdx.x & 31;
  const int t0 = wv * 2;
  const int mt = t0 >> 2, ntA = t0 & 3, ntB = ntA + 1;
  v8f acc0 = {}, acc1 = {};
#pragma unroll
  for (int kc = 0; kc < 8; ++kc) {
    v16bf a  = ld_tile_rowmaj(H + (m0 + mt * 16) * ND + kc * 32, ND, lane);
    v16bf bA = ld_tile_rowmaj(W + (n0 + ntA * 16) * ND + kc * 32, ND, lane);
    v16bf bB = ld_tile_rowmaj(W + (n0 + ntB * 16) * ND + kc * 32, ND, lane);
    acc0 = wmma_bf16(a, bA, acc0);
    acc1 = wmma_bf16(a, bB, acc1);
  }
  const int hf = lane >> 4, cl = lane & 15;
#pragma unroll
  for (int r = 0; r < 8; ++r) {
    const int row = m0 + mt * 16 + r + 8 * hf;  // = b*S + s
    const int b = row >> 10, s = row & (NS - 1);
    vT[(b * ND + n0 + ntA * 16 + cl) * NS + s] = (__bf16)acc0[r];
    vT[(b * ND + n0 + ntB * 16 + cl) * NS + s] = (__bf16)acc1[r];
  }
}

// ---------------------------------------------------------------------------
// Flash attention: grid (S/64, B), block 256 (8 waves).
// Per block: 64 query rows, online softmax over 16 key tiles of 64.
// V is consumed from vT[b][n][s], so the P@V B-operand is row-major.
// LDS: Q 32K + S 16K + P 8K + m/l/alpha 768B = ~57K.
// ---------------------------------------------------------------------------
__global__ void k_attn(const __bf16* __restrict__ q, const __bf16* __restrict__ k,
                       const __bf16* __restrict__ vT, float* __restrict__ out) {
  extern __shared__ char smem[];
  __bf16* Qs  = (__bf16*)smem;                              // [64][256]
  float*  Ss  = (float*)(smem + 32768);                     // [64][64]
  __bf16* Ps  = (__bf16*)(smem + 32768 + 16384);            // [64][64]
  float*  mrw = (float*)(smem + 32768 + 16384 + 8192);      // [64]
  float*  lrw = mrw + 64;
  float*  arw = mrw + 128;

  const int b = blockIdx.y, q0 = blockIdx.x * 64;
  const int tid = threadIdx.x, wv = tid >> 5, lane = tid & 31;
  const int hf = lane >> 4, cl = lane & 15;

  const __bf16* qb  = q  + (size_t)(b * NS + q0) * ND;
  const __bf16* kb  = k  + (size_t)b * NS * ND;
  const __bf16* vtb = vT + (size_t)b * ND * NS;

  { // stage Q tile into LDS as dwords
    const uint32_t* s32 = (const uint32_t*)qb;
    uint32_t* d32 = (uint32_t*)Qs;
    for (int u = tid; u < 64 * (ND / 2); u += 256) d32[u] = s32[u];
  }
  if (tid < 64) { mrw[tid] = -1e30f; lrw[tid] = 0.0f; }

  v8f O[4][2];
#pragma unroll
  for (int a = 0; a < 4; ++a)
#pragma unroll
    for (int c = 0; c < 2; ++c) O[a][c] = (v8f){};
  __syncthreads();

  const int t0 = wv * 2;
  const int smt = t0 >> 2, sntA = t0 & 3, sntB = sntA + 1;

  for (int j0 = 0; j0 < NS; j0 += 64) {
    // ---- scores S = (Q K^T) / 16, 16 tiles over 8 waves ----
    v8f s0 = {}, s1 = {};
#pragma unroll
    for (int kc = 0; kc < 8; ++kc) {
      v16bf a  = ld_tile_rowmaj(Qs + smt * 16 * ND + kc * 32, ND, lane);
      v16bf bA = ld_tile_rowmaj(kb + (j0 + sntA * 16) * ND + kc * 32, ND, lane);
      v16bf bB = ld_tile_rowmaj(kb + (j0 + sntB * 16) * ND + kc * 32, ND, lane);
      s0 = wmma_bf16(a, bA, s0);
      s1 = wmma_bf16(a, bB, s1);
    }
#pragma unroll
    for (int r = 0; r < 8; ++r) {
      const int row = smt * 16 + r + 8 * hf;
      Ss[row * 64 + sntA * 16 + cl] = s0[r] * 0.0625f;
      Ss[row * 64 + sntB * 16 + cl] = s1[r] * 0.0625f;
    }
    __syncthreads();

    // ---- online softmax (one thread per query row) ----
    if (tid < 64) {
      const float mo = mrw[tid];
      float mx = mo;
      for (int c = 0; c < 64; ++c) mx = fmaxf(mx, Ss[tid * 64 + c]);
      const float al = __expf(mo - mx);
      float ls = lrw[tid] * al;
      for (int c = 0; c < 64; ++c) {
        const float p = __expf(Ss[tid * 64 + c] - mx);
        ls += p;
        Ps[tid * 64 + c] = (__bf16)p;
      }
      mrw[tid] = mx; lrw[tid] = ls; arw[tid] = al;
    }
    __syncthreads();

    // ---- rescale O and accumulate P @ V (B from vT, row-major loads) ----
#pragma unroll
    for (int mt = 0; mt < 4; ++mt) {
      v16bf a0 = ld_tile_rowmaj(Ps + mt * 16 * 64, 64, lane);
      v16bf a1 = ld_tile_rowmaj(Ps + mt * 16 * 64 + 32, 64, lane);
#pragma unroll
      for (int nn = 0; nn < 2; ++nn) {
        const int nt = wv * 2 + nn;
#pragma unroll
        for (int r = 0; r < 8; ++r) O[mt][nn][r] *= arw[mt * 16 + r + 8 * hf];
        v16bf b0 = ld_tile_rowmaj(vtb + (nt * 16) * NS + j0, NS, lane);
        v16bf b1 = ld_tile_rowmaj(vtb + (nt * 16) * NS + j0 + 32, NS, lane);
        O[mt][nn] = wmma_bf16(a0, b0, O[mt][nn]);
        O[mt][nn] = wmma_bf16(a1, b1, O[mt][nn]);
      }
    }
    __syncthreads();
  }

  // ---- epilogue: O / l ----
#pragma unroll
  for (int mt = 0; mt < 4; ++mt)
#pragma unroll
    for (int nn = 0; nn < 2; ++nn) {
      const int nt = wv * 2 + nn;
#pragma unroll
      for (int r = 0; r < 8; ++r) {
        const int row = mt * 16 + r + 8 * hf;
        const float inv = __frcp_rn(lrw[row]);
        out[(size_t)(b * NS + q0 + row) * ND + nt * 16 + cl] =
            O[mt][nn][r] * inv;
      }
    }
}

// ---------------------------------------------------------------------------
extern "C" void kernel_launch(void* const* d_in, const int* in_sizes, int n_in,
                              void* d_out, int out_size, void* d_ws, size_t ws_size,
                              hipStream_t stream) {
  (void)in_sizes; (void)n_in; (void)out_size; (void)ws_size;
  const float* x      = (const float*)d_in[0];
  const float* fw_Wih = (const float*)d_in[1];
  const float* fw_Whh = (const float*)d_in[2];
  const float* fw_bih = (const float*)d_in[3];
  const float* fw_bhh = (const float*)d_in[4];
  const float* bw_Wih = (const float*)d_in[5];
  const float* bw_Whh = (const float*)d_in[6];
  const float* bw_bih = (const float*)d_in[7];
  const float* bw_bhh = (const float*)d_in[8];
  const float* Wq     = (const float*)d_in[9];
  const float* Wk     = (const float*)d_in[10];
  const float* Wv     = (const float*)d_in[11];

  char* ws = (char*)d_ws;
  float*  xg_fw = (float*)(ws + OFF_XG_FW);
  float*  xg_bw = (float*)(ws + OFF_XG_BW);
  __bf16* hb    = (__bf16*)(ws + OFF_H);
  __bf16* xbf   = (__bf16*)(ws + OFF_XBF);
  __bf16* xrev  = (__bf16*)(ws + OFF_XREV);
  __bf16* wihF  = (__bf16*)(ws + OFF_WTS + 0x00000);
  __bf16* whhF  = (__bf16*)(ws + OFF_WTS + 0x20000);
  __bf16* wihB  = (__bf16*)(ws + OFF_WTS + 0x40000);
  __bf16* whhB  = (__bf16*)(ws + OFF_WTS + 0x60000);
  __bf16* wqb   = (__bf16*)(ws + OFF_WTS + 0x80000);
  __bf16* wkb   = (__bf16*)(ws + OFF_WTS + 0xA0000);
  __bf16* wvb   = (__bf16*)(ws + OFF_WTS + 0xC0000);
  float*  biasF = (float*)(ws + OFF_WTS + 0xE0000);
  float*  biasB = (float*)(ws + OFF_WTS + 0xE0800);
  __bf16* qb    = (__bf16*)(ws + OFF_Q);
  __bf16* kbp   = (__bf16*)(ws + OFF_K);
  __bf16* vTp   = (__bf16*)(ws + OFF_VT);

  // --- pack to bf16 ---
  const int nx = NB * NS * NI;             // 4,194,304
  k_cvt_bf16<<<(nx + 255) / 256, 256, 0, stream>>>(x, xbf, nx);
  k_rev_x<<<(nx + 255) / 256, 256, 0, stream>>>(xbf, xrev);
  k_cvt_bf16<<<256, 256, 0, stream>>>(fw_Wih, wihF, NG * NI);
  k_cvt_bf16<<<256, 256, 0, stream>>>(fw_Whh, whhF, NG * NH);
  k_cvt_bf16<<<256, 256, 0, stream>>>(bw_Wih, wihB, NG * NI);
  k_cvt_bf16<<<256, 256, 0, stream>>>(bw_Whh, whhB, NG * NH);
  k_cvt_bf16<<<256, 256, 0, stream>>>(Wq, wqb, ND * ND);
  k_cvt_bf16<<<256, 256, 0, stream>>>(Wk, wkb, ND * ND);
  k_cvt_bf16<<<256, 256, 0, stream>>>(Wv, wvb, ND * ND);
  k_bias_add<<<2, 256, 0, stream>>>(fw_bih, fw_bhh, biasF, NG);
  k_bias_add<<<2, 256, 0, stream>>>(bw_bih, bw_bhh, biasB, NG);

  // --- input projections (WMMA) ---
  dim3 gxg(NB * NS / 64, NG / 64);
  k_gemm_xg<<<gxg, 256, 0, stream>>>(xbf, wihF, biasF, xg_fw);
  k_gemm_xg<<<gxg, 256, 0, stream>>>(xrev, wihB, biasB, xg_bw);

  // --- recurrence (persistent, 88KB LDS) ---
  hipFuncSetAttribute((const void*)k_lstm,
                      hipFuncAttributeMaxDynamicSharedMemorySize, 90112);
  k_lstm<<<2, 1024, 90112, stream>>>(xg_fw, xg_bw, whhF, whhB, hb);

  // --- QKV projections (WMMA); V written transposed for attention ---
  dim3 gqkv(NB * NS / 64, ND / 64);
  k_gemm_qk<<<gqkv, 256, 0, stream>>>(hb, wqb, qb);
  k_gemm_qk<<<gqkv, 256, 0, stream>>>(hb, wkb, kbp);
  k_gemm_vT<<<gqkv, 256, 0, stream>>>(hb, wvb, vTp);

  // --- flash attention ---
  hipFuncSetAttribute((const void*)k_attn,
                      hipFuncAttributeMaxDynamicSharedMemorySize, 58368);
  dim3 gat(NS / 64, NB);
  k_attn<<<gat, 256, 58368, stream>>>(qb, kbp, vTp, (float*)d_out);
}
